// GNNBackbone_2044404433058
// MI455X (gfx1250) — matrix operand order repaired and checked by reading, not compiled, not run on hardware
//
#include <hip/hip_runtime.h>
#include <hip/hip_bf16.h>

#define N_NODES   50000
#define N_EDGES   800000
#define NUM_GRAPHS 64
#define IN_C      800
#define HID_C     256

typedef __attribute__((ext_vector_type(16))) __bf16 v16bf;
typedef __attribute__((ext_vector_type(8)))  float  v8f;

__device__ __forceinline__ unsigned short f2bf(float f) {
  unsigned int u = __float_as_uint(f);
  unsigned int r = u + 0x7FFFu + ((u >> 16) & 1u);   // round-to-nearest-even
  return (unsigned short)(r >> 16);
}

// ---------------- conversion kernels ----------------

__global__ void cvt_bf16_kernel(const float* __restrict__ in,
                                unsigned short* __restrict__ out, int n) {
  int i = blockIdx.x * blockDim.x + threadIdx.x;
  if (i < n) out[i] = f2bf(in[i]);
}

// W [K x 256] fp32 -> Wt [256 x K] bf16 (transposed so B-tiles are contiguous per column)
__global__ void cvt_w_transpose_kernel(const float* __restrict__ W,
                                       unsigned short* __restrict__ Wt, int K) {
  int idx = blockIdx.x * blockDim.x + threadIdx.x;
  if (idx >= K * 256) return;
  int k = idx >> 8;
  int n = idx & 255;
  Wt[(size_t)n * K + k] = f2bf(W[(size_t)k * 256 + n]);
}

// ---------------- graph kernels ----------------

__global__ void deg_kernel(const int* __restrict__ dst, float* __restrict__ deg, int nE) {
  int i = blockIdx.x * blockDim.x + threadIdx.x;
  if (i < nE) atomicAdd(&deg[dst[i]], 1.0f);
}

// 64 threads per edge: gather float4 from Y[src], atomic-add into Agg[dst] (L2-resident)
__global__ void scatter_kernel(const int* __restrict__ src, const int* __restrict__ dst,
                               const float* __restrict__ Y, float* __restrict__ Agg, int nE) {
  long long t = (long long)blockIdx.x * blockDim.x + threadIdx.x;
  int e = (int)(t >> 6);
  if (e >= nE) return;
  int c = ((int)t & 63) << 2;
  int s = src[e];
  int d = dst[e];
  const float4 v = *(const float4*)(Y + (size_t)s * 256 + c);
  float* o = Agg + (size_t)d * 256 + c;
  atomicAdd(o + 0, v.x);
  atomicAdd(o + 1, v.y);
  atomicAdd(o + 2, v.z);
  atomicAdd(o + 3, v.w);
}

// h = Agg/max(deg,1) + bias + Yr ; optional relu ; write fp32 (final layer) / bf16 (next layer)
__global__ void finalize_kernel(const float* __restrict__ Agg, const float* __restrict__ Yr,
                                const float* __restrict__ bias, const float* __restrict__ deg,
                                float* __restrict__ Hf, unsigned short* __restrict__ Hb,
                                int do_relu, int M) {
  int idx = blockIdx.x * blockDim.x + threadIdx.x;
  if (idx >= M * 256) return;
  int i = idx >> 8;
  int c = idx & 255;
  float h = Agg[idx] / fmaxf(deg[i], 1.0f) + bias[c] + Yr[idx];
  if (do_relu) h = fmaxf(h, 0.0f);
  if (Hf) Hf[idx] = h;
  if (Hb) Hb[idx] = f2bf(h);
}

// ---------------- WMMA GEMM: C[M x 256] = A[M x K] (bf16) * B[K x 256] (bf16, pre-transposed) ----
// Each wave computes a 32x64 output tile (2 M-subtiles x 4 N-subtiles = 8 WMMAs per K-step),
// so each B-tile load is reused by two WMMAs. A buffer is padded by 32 rows so the upper
// half-tile loads are always in-bounds; its stores are guarded wave-uniformly.

__global__ __launch_bounds__(256) void gemm_bf16_wmma_kernel(
    const unsigned short* __restrict__ A,   // [>=M x K] row-major bf16 (padded to 32-row multiple)
    const unsigned short* __restrict__ Bt,  // [256 x K] row-major bf16 (== B transposed)
    float* __restrict__ C,                  // [M x 256] fp32
    int M, int K) {
  const int wave = threadIdx.x >> 5;
  const int lane = threadIdx.x & 31;
  const int half = lane >> 4;   // lanes 0-15 vs 16-31
  const int lm   = lane & 15;
  const int mt = blockIdx.x * 8 + wave;     // 32-row tile index
  if (mt * 32 >= M) return;                 // wave-uniform guard (EXEC all-1s for WMMA)
  const int m0 = mt * 32;
  const bool full = (m0 + 16) < M;          // wave-uniform: is upper 16-row subtile valid?
  const int n0 = blockIdx.y * 64;

  union V16 { v16bf v; uint4 q[2]; };

  v8f acc[2][4] = {};

  // ISA 16-bit A 16x32 layout: lanes0-15 K 0-7 / 16-23, lanes16-31 K 8-15 / 24-31
  const unsigned short* arow0 = A + (size_t)(m0 + lm) * K + half * 8;
  const unsigned short* arow1 = arow0 + (size_t)16 * K;
  // B 32x16 layout: lane = column, lanes0-15 K 0-7 / 16-23, lanes16-31 K 8-15 / 24-31
  const unsigned short* brow0 = Bt + (size_t)(n0 + lm) * K + half * 8;
  const size_t bstride = (size_t)16 * K;    // 16 columns to the next n-subtile

  for (int k0 = 0; k0 < K; k0 += 32) {
    V16 a0, a1;
    a0.q[0] = *(const uint4*)(arow0 + k0);        // K = k0 + half*8 + [0..7]
    a0.q[1] = *(const uint4*)(arow0 + k0 + 16);   // K = k0 + 16 + half*8 + [0..7]
    a1.q[0] = *(const uint4*)(arow1 + k0);
    a1.q[1] = *(const uint4*)(arow1 + k0 + 16);
#pragma unroll
    for (int j = 0; j < 4; ++j) {
      V16 b;
      const unsigned short* bp = brow0 + (size_t)j * bstride + k0;
      b.q[0] = *(const uint4*)(bp);               // K = k0 + half*8 + [0..7]
      b.q[1] = *(const uint4*)(bp + 16);          // K = k0 + 16 + half*8 + [0..7]
      acc[0][j] = __builtin_amdgcn_wmma_f32_16x16x32_bf16(
          false, a0.v, false, b.v, (short)0, acc[0][j], false, false);
      acc[1][j] = __builtin_amdgcn_wmma_f32_16x16x32_bf16(
          false, a1.v, false, b.v, (short)0, acc[1][j], false, false);
    }
  }

  // C/D layout: VGPR r -> row base + half*8 + r, lane lm -> column
#pragma unroll
  for (int j = 0; j < 4; ++j) {
    int col = n0 + j * 16 + lm;
#pragma unroll
    for (int r = 0; r < 8; ++r) {
      int row = m0 + half * 8 + r;
      C[(size_t)row * 256 + col] = acc[0][j][r];
    }
  }
  if (full) {
#pragma unroll
    for (int j = 0; j < 4; ++j) {
      int col = n0 + j * 16 + lm;
#pragma unroll
      for (int r = 0; r < 8; ++r) {
        int row = m0 + 16 + half * 8 + r;
        C[(size_t)row * 256 + col] = acc[1][j][r];
      }
    }
  }
}

// ---------------- pooling ----------------

__global__ void pool_cnt_kernel(const int* __restrict__ batch, float* __restrict__ cnt, int M) {
  int i = blockIdx.x * blockDim.x + threadIdx.x;
  if (i < M) atomicAdd(&cnt[batch[i]], 1.0f);
}

__global__ void pool_sum_kernel(const float* __restrict__ xn, const int* __restrict__ batch,
                                float* __restrict__ gsum) {
  int i = blockIdx.x;       // node
  int c = threadIdx.x;      // channel (256)
  int g = batch[i];
  atomicAdd(&gsum[(size_t)g * 256 + c], xn[(size_t)i * 256 + c]);
}

__global__ void pool_div_kernel(float* __restrict__ gsum, const float* __restrict__ cnt) {
  int idx = blockIdx.x * blockDim.x + threadIdx.x;
  if (idx < NUM_GRAPHS * 256) gsum[idx] /= fmaxf(cnt[idx >> 8], 1.0f);
}

// ---------------- host driver ----------------

extern "C" void kernel_launch(void* const* d_in, const int* in_sizes, int n_in,
                              void* d_out, int out_size, void* d_ws, size_t ws_size,
                              hipStream_t stream) {
  const float* x     = (const float*)d_in[0];
  const int*   edge  = (const int*)d_in[1];
  const int*   batch = (const int*)d_in[2];
  const int*   src = edge;
  const int*   dst = edge + N_EDGES;
  const float* Wl[4] = {(const float*)d_in[3], (const float*)d_in[6],
                        (const float*)d_in[9], (const float*)d_in[12]};
  const float* bb[4] = {(const float*)d_in[4], (const float*)d_in[7],
                        (const float*)d_in[10], (const float*)d_in[13]};
  const float* Wr[4] = {(const float*)d_in[5], (const float*)d_in[8],
                        (const float*)d_in[11], (const float*)d_in[14]};

  char* ws = (char*)d_ws;
  size_t off = 0;
  auto take = [&](size_t bytes) {
    char* p = ws + off;
    off += (bytes + 255) & ~(size_t)255;
    return p;
  };
  float*          deg = (float*)take((size_t)N_NODES * 4);
  float*          cnt = (float*)take((size_t)NUM_GRAPHS * 4);
  // +32 padded rows so the 32-row GEMM tiles can always load the upper half-tile
  unsigned short* Xb  = (unsigned short*)take((size_t)(N_NODES + 32) * IN_C * 2);
  unsigned short* Wlt = (unsigned short*)take((size_t)256 * IN_C * 2);
  unsigned short* Wrt = (unsigned short*)take((size_t)256 * IN_C * 2);
  float*          Yl  = (float*)take((size_t)N_NODES * 256 * 4);
  float*          Yr  = (float*)take((size_t)N_NODES * 256 * 4);
  float*          Agg = (float*)take((size_t)N_NODES * 256 * 4);

  float* out    = (float*)d_out;
  float* gsum   = out;                        // x_graph [64 x 256]
  float* xnodes = out + NUM_GRAPHS * 256;     // x_nodes [50000 x 256]

  // degrees (shared by all layers)
  (void)hipMemsetAsync(deg, 0, (size_t)N_NODES * 4, stream);
  deg_kernel<<<(N_EDGES + 255) / 256, 256, 0, stream>>>(dst, deg, N_EDGES);

  // convert input features once to bf16
  {
    int n = N_NODES * IN_C;
    cvt_bf16_kernel<<<(n + 255) / 256, 256, 0, stream>>>(x, Xb, n);
  }

  const int mtiles32 = (N_NODES + 31) / 32;   // 1563
  for (int L = 0; L < 4; ++L) {
    const int K = (L == 0) ? IN_C : HID_C;
    const int wn = K * 256;
    cvt_w_transpose_kernel<<<(wn + 255) / 256, 256, 0, stream>>>(Wl[L], Wlt, K);
    cvt_w_transpose_kernel<<<(wn + 255) / 256, 256, 0, stream>>>(Wr[L], Wrt, K);

    dim3 grid((mtiles32 + 7) / 8, 4);
    gemm_bf16_wmma_kernel<<<grid, 256, 0, stream>>>(Xb, Wlt, Yl, N_NODES, K);
    gemm_bf16_wmma_kernel<<<grid, 256, 0, stream>>>(Xb, Wrt, Yr, N_NODES, K);

    (void)hipMemsetAsync(Agg, 0, (size_t)N_NODES * 256 * 4, stream);
    {
      long long t = (long long)N_EDGES * 64;
      int blocks = (int)((t + 255) / 256);
      scatter_kernel<<<blocks, 256, 0, stream>>>(src, dst, Yl, Agg, N_EDGES);
    }

    const int total = N_NODES * 256;
    if (L < 3) {
      // write next-layer bf16 input in place over Xb (no longer needed this layer)
      finalize_kernel<<<(total + 255) / 256, 256, 0, stream>>>(
          Agg, Yr, bb[L], deg, nullptr, Xb, 1, N_NODES);
    } else {
      finalize_kernel<<<(total + 255) / 256, 256, 0, stream>>>(
          Agg, Yr, bb[L], deg, xnodes, nullptr, 0, N_NODES);
    }
  }

  // global mean pool
  (void)hipMemsetAsync(gsum, 0, (size_t)NUM_GRAPHS * 256 * 4, stream);
  (void)hipMemsetAsync(cnt, 0, (size_t)NUM_GRAPHS * 4, stream);
  pool_cnt_kernel<<<(N_NODES + 255) / 256, 256, 0, stream>>>(batch, cnt, N_NODES);
  pool_sum_kernel<<<N_NODES, 256, 0, stream>>>(xnodes, batch, gsum);
  pool_div_kernel<<<(NUM_GRAPHS * 256 + 255) / 256, 256, 0, stream>>>(gsum, cnt);
}
